// MultiHeadAttention_43628277793601
// MI455X (gfx1250) — matrix operand order repaired and checked by reading, not compiled, not run on hardware
//
#include <hip/hip_runtime.h>

typedef __bf16 bf16_t;
typedef __attribute__((ext_vector_type(16))) __bf16 v16bf;
typedef __attribute__((ext_vector_type(8)))  __bf16 v8bf;
typedef __attribute__((ext_vector_type(8)))  float  v8f;
typedef __attribute__((ext_vector_type(4)))  int    i32x4;

#define WMMA_BF16(a, b, c) \
    __builtin_amdgcn_wmma_f32_16x16x32_bf16(false, (a), false, (b), (short)0, (c), false, false)

// Problem constants
#define BATCH 4
#define SEQ   2048
#define DIM   1024
#define HEADS 16
#define DHEAD 64
#define ROWS  (BATCH * SEQ)     // 8192
#define NEGBIG (-1.0e9f)

// CDNA5 async global->LDS copy (ASYNCcnt-tracked). Guarded so the file still
// compiles if the toolchain lacks the builtin (falls back to reg-staged copy).
#if defined(__has_builtin)
#  if __has_builtin(__builtin_amdgcn_global_load_async_to_lds_b128) && \
      __has_builtin(__builtin_amdgcn_s_wait_asynccnt)
#    define HAVE_ASYNC_LDS 1
#  endif
#endif
#ifndef HAVE_ASYNC_LDS
#  define HAVE_ASYNC_LDS 0
#endif

#define AS1 __attribute__((address_space(1)))
#define AS3 __attribute__((address_space(3)))

__device__ __forceinline__ v8f v8f_zero() {
    v8f z;
#pragma unroll
    for (int i = 0; i < 8; ++i) z[i] = 0.f;
    return z;
}

// ---------------------------------------------------------------------------
// Kernel 0: convert + transpose weights f32 [in][out] -> bf16 [out][in]
// ---------------------------------------------------------------------------
__global__ __launch_bounds__(256) void prep_weights_kernel(
    const float* __restrict__ Wq, const float* __restrict__ Wk,
    const float* __restrict__ Wv, const float* __restrict__ Wo,
    bf16_t* __restrict__ WqT, bf16_t* __restrict__ WkT,
    bf16_t* __restrict__ WvT, bf16_t* __restrict__ WoT)
{
    const size_t i = (size_t)blockIdx.x * 256 + threadIdx.x; // 4M elements
    const int selw = (int)(i >> 20);
    const int idx  = (int)(i & 1048575u);
    const int kk = idx >> 10;      // input row
    const int n  = idx & 1023;     // output col
    const float* W  = (selw == 0) ? Wq : (selw == 1) ? Wk : (selw == 2) ? Wv : Wo;
    bf16_t*      Wt = (selw == 0) ? WqT : (selw == 1) ? WkT : (selw == 2) ? WvT : WoT;
    Wt[(size_t)n * DIM + kk] = (bf16_t)W[idx];
}

// ---------------------------------------------------------------------------
// Kernel 1: fused QKV projection.  X[8192,1024] @ W + b
//   Q,K -> bf16 [B,H,T,DH];  V -> bf16 [B,H,DH,T] (transposed for attention)
// One wave computes a 16x64 output tile; 3*512*16 = 24576 waves.
// ---------------------------------------------------------------------------
__global__ __launch_bounds__(256) void qkv_proj_kernel(
    const float* __restrict__ qin, const float* __restrict__ kin, const float* __restrict__ vin,
    const bf16_t* __restrict__ WqT, const bf16_t* __restrict__ WkT, const bf16_t* __restrict__ WvT,
    const float* __restrict__ bq, const float* __restrict__ bk, const float* __restrict__ bv,
    bf16_t* __restrict__ Qh, bf16_t* __restrict__ Kh, bf16_t* __restrict__ VhT)
{
    const int wid = blockIdx.x * 8 + (threadIdx.x >> 5);
    const int sel = wid >> 13;            // / 8192
    const int rem = wid & 8191;
    const int mt  = rem >> 4;             // 0..511 (16-row tile)
    const int ng  = rem & 15;             // 0..15  (64-col group)

    const float*  X    = (sel == 0) ? qin : (sel == 1) ? kin : vin;
    const bf16_t* Wt   = (sel == 0) ? WqT : (sel == 1) ? WkT : WvT;
    const float*  bias = (sel == 0) ? bq  : (sel == 1) ? bk  : bv;
    bf16_t*       Out  = (sel == 0) ? Qh  : (sel == 1) ? Kh  : VhT;

    const int lane = threadIdx.x & 31;
    const int hl   = lane >> 4;           // half-wave
    const int mr   = lane & 15;
    const int m0   = mt * 16;
    const int n0   = ng * 64;

    v8f acc[4];
#pragma unroll
    for (int t = 0; t < 4; ++t) acc[t] = v8f_zero();

    for (int k0 = 0; k0 < DIM; k0 += 32) {
        // A fragment: lane mr holds row m0+mr; elements 0..15 -> k = k0+hl*16+{0..15}
        const float* arow = X + (size_t)(m0 + mr) * DIM + k0 + hl * 16;
        const float4 a0 = *(const float4*)(arow);
        const float4 a1 = *(const float4*)(arow + 4);
        const float4 a2 = *(const float4*)(arow + 8);
        const float4 a3 = *(const float4*)(arow + 12);
        v16bf af;
        af[0]  = (bf16_t)a0.x; af[1]  = (bf16_t)a0.y; af[2]  = (bf16_t)a0.z; af[3]  = (bf16_t)a0.w;
        af[4]  = (bf16_t)a1.x; af[5]  = (bf16_t)a1.y; af[6]  = (bf16_t)a1.z; af[7]  = (bf16_t)a1.w;
        af[8]  = (bf16_t)a2.x; af[9]  = (bf16_t)a2.y; af[10] = (bf16_t)a2.z; af[11] = (bf16_t)a2.w;
        af[12] = (bf16_t)a3.x; af[13] = (bf16_t)a3.y; af[14] = (bf16_t)a3.z; af[15] = (bf16_t)a3.w;

#pragma unroll
        for (int t = 0; t < 4; ++t) {
            // B fragment: lane col n0+t*16+mr; k = k0 + hl*16 + e (contiguous in Wt row)
            const v16bf wf = *(const v16bf*)(Wt + (size_t)(n0 + t * 16 + mr) * DIM + k0 + hl * 16);
            acc[t] = WMMA_BF16(af, wf, acc[t]);
        }
    }

#pragma unroll
    for (int t = 0; t < 4; ++t) {
        const int n = n0 + t * 16 + mr;
        const int h = n >> 6;
        const int d = n & 63;
        const float bn = bias[n];
#pragma unroll
        for (int r = 0; r < 8; ++r) {
            const int m  = m0 + r + hl * 8;
            const int b  = m >> 11;         // / SEQ
            const int tq = m & 2047;
            const bf16_t val = (bf16_t)(acc[t][r] + bn);
            if (sel < 2)
                Out[(((size_t)b * HEADS + h) * SEQ + tq) * DHEAD + d] = val;      // [B,H,T,DH]
            else
                Out[(((size_t)b * HEADS + h) * DHEAD + d) * SEQ + tq] = val;      // [B,H,DH,T]
        }
    }
}

// ---------------------------------------------------------------------------
// Kernel 2: causal + context-masked flash attention per (b,h).
// Workgroup = 8 waves; wave owns 16 query rows; key blocks of 32.
// K/V tiles double-buffered in LDS via async global->LDS copies: the copies
// for block i+1 are issued before computing block i, and the ASYNCcnt wait is
// deferred to the end of the iteration (memory/compute overlap).
// ---------------------------------------------------------------------------
__global__ __launch_bounds__(256) void flash_attn_kernel(
    const bf16_t* __restrict__ Qh, const bf16_t* __restrict__ Kh,
    const bf16_t* __restrict__ VhT, const unsigned char* __restrict__ cmask,
    bf16_t* __restrict__ Oat)   // [B,T,H*DH] bf16
{
    __shared__ bf16_t Ksh[2][32][64 + 8];   // key blocks, row-major [key][d]  (+pad)
    __shared__ bf16_t Vsh[2][64][32 + 8];   // value blocks, [d][key]          (+pad)
    __shared__ bf16_t psh[8][16][32 + 8];   // per-wave P bounce (C-layout -> A-layout)

    const int wg   = blockIdx.x;
    const int qt   = wg & 15;            // T/128 tiles
    const int bh   = wg >> 4;
    const int b    = bh >> 4;
    const int h    = bh & 15;
    const int wave = threadIdx.x >> 5;
    const int lane = threadIdx.x & 31;
    const int hl   = lane >> 4;
    const int mr   = lane & 15;
    const int q0   = qt * 128 + wave * 16;
    const int tid  = threadIdx.x;

    const size_t bhbase = ((size_t)b * HEADS + h) * (size_t)SEQ * DHEAD;
    const bf16_t* Q  = Qh  + bhbase;     // [T][64]
    const bf16_t* K  = Kh  + bhbase;     // [T][64]
    const bf16_t* Vt = VhT + bhbase;     // [64][T]
    const unsigned char* cm = cmask + (size_t)b * SEQ;

    // Q fragments for DH=64: two K-steps of 32
    v16bf qf[2];
#pragma unroll
    for (int s = 0; s < 2; ++s) {
        const bf16_t* qr = Q + (size_t)(q0 + mr) * DHEAD + s * 32 + hl * 16;
        const v8bf lo = *(const v8bf*)(qr);
        const v8bf hi = *(const v8bf*)(qr + 8);
#pragma unroll
        for (int e = 0; e < 8; ++e) { qf[s][e] = lo[e]; qf[s][8 + e] = hi[e]; }
    }

    float rmax[8], rsum[8];
#pragma unroll
    for (int r = 0; r < 8; ++r) { rmax[r] = -3.0e38f; rsum[r] = 0.f; }
    v8f oacc[4];
#pragma unroll
    for (int t = 0; t < 4; ++t) oacc[t] = v8f_zero();

    // cooperative staging indices (256 threads, 16B chunks)
    const int krow = tid >> 3, kch = tid & 7;   // K tile: 32 rows x 128B
    const int vrow = tid >> 2, vch = tid & 3;   // V tile: 64 rows x 64B

    const int kend_wave = q0 + 16;              // this wave's causal limit
    const int kend_wg   = qt * 128 + 128;       // workgroup-uniform loop bound

    // issue K/V tile copies for key block kb into LDS buffer `buf`
    auto stage = [&](int kb, int buf) {
        const bf16_t* gk = K  + (size_t)(kb + krow) * DHEAD + kch * 8;
        const bf16_t* gv = Vt + (size_t)vrow * SEQ + kb + vch * 8;
        bf16_t* lk = &Ksh[buf][krow][kch * 8];
        bf16_t* lv = &Vsh[buf][vrow][vch * 8];
#if HAVE_ASYNC_LDS
        __builtin_amdgcn_global_load_async_to_lds_b128(
            (AS1 i32x4*)gk, (AS3 i32x4*)lk, 0, 0);
        __builtin_amdgcn_global_load_async_to_lds_b128(
            (AS1 i32x4*)gv, (AS3 i32x4*)lv, 0, 0);
#else
        *(float4*)lk = *(const float4*)gk;
        *(float4*)lv = *(const float4*)gv;
#endif
    };

    // prologue: stage block 0, wait, sync
    stage(0, 0);
#if HAVE_ASYNC_LDS
    __builtin_amdgcn_s_wait_asynccnt(0);
#endif
    __syncthreads();

    for (int kb = 0; kb < kend_wg; kb += 32) {
        const int cur = (kb >> 5) & 1;
        // prefetch next key block into the other buffer (deferred wait)
        if (kb + 32 < kend_wg) stage(kb + 32, cur ^ 1);

        if (kb < kend_wave) {
            // ---- S = Q * K^T for 16x32 key block (two 16x16 N-tiles) ----
            v8f sf[2];
#pragma unroll
            for (int nt = 0; nt < 2; ++nt) {
                sf[nt] = v8f_zero();
                const bf16_t* krp = &Ksh[cur][nt * 16 + mr][0];
#pragma unroll
                for (int s = 0; s < 2; ++s) {
                    const v16bf kf = *(const v16bf*)(krp + s * 32 + hl * 16);
                    sf[nt] = WMMA_BF16(qf[s], kf, sf[nt]);
                }
            }
            // ---- scale + mask, per-lane row stats ----
            float pv[2][8], lmax[8];
#pragma unroll
            for (int r = 0; r < 8; ++r) lmax[r] = -3.0e38f;
#pragma unroll
            for (int nt = 0; nt < 2; ++nt) {
                const int tk = kb + nt * 16 + mr;
                const bool keep = cm[tk] != 0;
#pragma unroll
                for (int r = 0; r < 8; ++r) {
                    const int tq = q0 + r + hl * 8;
                    float sval = sf[nt][r] * 0.125f;          // DH^-0.5
                    if (!keep || tk > tq) sval = NEGBIG;
                    pv[nt][r] = sval;
                    lmax[r] = fmaxf(lmax[r], sval);
                }
            }
            // row max across the 16 lanes sharing each row
#pragma unroll
            for (int off = 1; off < 16; off <<= 1)
#pragma unroll
                for (int r = 0; r < 8; ++r)
                    lmax[r] = fmaxf(lmax[r], __shfl_xor(lmax[r], off, 32));
            // online-softmax update
            float fac[8];
#pragma unroll
            for (int r = 0; r < 8; ++r) {
                const float nm = fmaxf(rmax[r], lmax[r]);
                fac[r] = __expf(rmax[r] - nm);
                rmax[r] = nm;
            }
            float psum[8];
#pragma unroll
            for (int r = 0; r < 8; ++r) psum[r] = 0.f;
#pragma unroll
            for (int nt = 0; nt < 2; ++nt)
#pragma unroll
                for (int r = 0; r < 8; ++r) {
                    const float p = __expf(pv[nt][r] - rmax[r]);
                    psum[r] += p;
                    psh[wave][r + hl * 8][nt * 16 + mr] = (bf16_t)p;   // row-major P tile
                }
#pragma unroll
            for (int off = 1; off < 16; off <<= 1)
#pragma unroll
                for (int r = 0; r < 8; ++r)
                    psum[r] += __shfl_xor(psum[r], off, 32);
#pragma unroll
            for (int r = 0; r < 8; ++r) rsum[r] = rsum[r] * fac[r] + psum[r];
#pragma unroll
            for (int t = 0; t < 4; ++t)
#pragma unroll
                for (int r = 0; r < 8; ++r) oacc[t][r] *= fac[r];

            // ---- re-stripe P into A-layout via LDS ----
            v16bf pf;
            {
                const bf16_t* prow = &psh[wave][mr][hl * 16];
#pragma unroll
                for (int e = 0; e < 8; ++e) { pf[e] = prow[e]; pf[8 + e] = prow[8 + e]; }
            }
            // ---- O += P * V : B-frag contiguous in Vsh [d][key] ----
#pragma unroll
            for (int t = 0; t < 4; ++t) {
                const v16bf vf = *(const v16bf*)(&Vsh[cur][t * 16 + mr][hl * 16]);
                oacc[t] = WMMA_BF16(pf, vf, oacc[t]);
            }
        }
        // ensure this wave's prefetch landed, then rendezvous before swapping
#if HAVE_ASYNC_LDS
        __builtin_amdgcn_s_wait_asynccnt(0);
#endif
        __syncthreads();
    }

    // epilogue: normalize and store [B,T,H*DH]
#pragma unroll
    for (int t = 0; t < 4; ++t)
#pragma unroll
        for (int r = 0; r < 8; ++r) {
            const int tq = q0 + r + hl * 8;
            const float o = oacc[t][r] / rsum[r];
            Oat[((size_t)b * SEQ + tq) * DIM + h * DHEAD + t * 16 + mr] = (bf16_t)o;
        }
}

// ---------------------------------------------------------------------------
// Kernel 3: output projection. Oat[8192,1024]bf16 @ WoT + b_o -> f32 out
// ---------------------------------------------------------------------------
__global__ __launch_bounds__(256) void out_proj_kernel(
    const bf16_t* __restrict__ A, const bf16_t* __restrict__ WoT,
    const float* __restrict__ bo, float* __restrict__ out)
{
    const int wid = blockIdx.x * 8 + (threadIdx.x >> 5);
    const int mt  = wid >> 4;
    const int ng  = wid & 15;
    const int lane = threadIdx.x & 31;
    const int hl   = lane >> 4;
    const int mr   = lane & 15;
    const int m0   = mt * 16;
    const int n0   = ng * 64;

    v8f acc[4];
#pragma unroll
    for (int t = 0; t < 4; ++t) acc[t] = v8f_zero();

    for (int k0 = 0; k0 < DIM; k0 += 32) {
        const bf16_t* arow = A + (size_t)(m0 + mr) * DIM + k0 + hl * 16;
        const v8bf lo = *(const v8bf*)(arow);
        const v8bf hi = *(const v8bf*)(arow + 8);
        v16bf af;
#pragma unroll
        for (int e = 0; e < 8; ++e) { af[e] = lo[e]; af[8 + e] = hi[e]; }
#pragma unroll
        for (int t = 0; t < 4; ++t) {
            const v16bf wf = *(const v16bf*)(WoT + (size_t)(n0 + t * 16 + mr) * DIM + k0 + hl * 16);
            acc[t] = WMMA_BF16(af, wf, acc[t]);
        }
    }

#pragma unroll
    for (int t = 0; t < 4; ++t) {
        const int n = n0 + t * 16 + mr;
        const float bn = bo[n];
#pragma unroll
        for (int r = 0; r < 8; ++r) {
            const int m = m0 + r + hl * 8;
            out[(size_t)m * DIM + n] = acc[t][r] + bn;
        }
    }
}

// ---------------------------------------------------------------------------
extern "C" void kernel_launch(void* const* d_in, const int* in_sizes, int n_in,
                              void* d_out, int out_size, void* d_ws, size_t ws_size,
                              hipStream_t stream) {
    (void)in_sizes; (void)n_in; (void)out_size; (void)ws_size;
    const float* q  = (const float*)d_in[0];
    const float* k  = (const float*)d_in[1];
    const float* v  = (const float*)d_in[2];
    const float* Wq = (const float*)d_in[3];
    const float* bq = (const float*)d_in[4];
    const float* Wk = (const float*)d_in[5];
    const float* bk = (const float*)d_in[6];
    const float* Wv = (const float*)d_in[7];
    const float* bv = (const float*)d_in[8];
    const float* Wo = (const float*)d_in[9];
    const float* bo = (const float*)d_in[10];
    // d_in[11] = causal mask (handled analytically)
    const unsigned char* ctx = (const unsigned char*)d_in[12];
    float* out = (float*)d_out;

    char* ws = (char*)d_ws;
    const size_t MB = 1024u * 1024u;
    bf16_t* Qh  = (bf16_t*)(ws);             // 16 MB  [B,H,T,DH]
    bf16_t* Kh  = (bf16_t*)(ws + 16 * MB);   // 16 MB  [B,H,T,DH]
    bf16_t* VhT = (bf16_t*)(ws + 32 * MB);   // 16 MB  [B,H,DH,T]
    bf16_t* Oat = (bf16_t*)(ws + 48 * MB);   // 16 MB  [B,T,H*DH]
    bf16_t* WqT = (bf16_t*)(ws + 64 * MB);   // 2 MB each, transposed bf16
    bf16_t* WkT = (bf16_t*)(ws + 66 * MB);
    bf16_t* WvT = (bf16_t*)(ws + 68 * MB);
    bf16_t* WoT = (bf16_t*)(ws + 70 * MB);

    prep_weights_kernel<<<16384, 256, 0, stream>>>(Wq, Wk, Wv, Wo, WqT, WkT, WvT, WoT);
    qkv_proj_kernel<<<3072, 256, 0, stream>>>(q, k, v, WqT, WkT, WvT, bq, bk, bv, Qh, Kh, VhT);
    flash_attn_kernel<<<1024, 256, 0, stream>>>(Qh, Kh, VhT, ctx, Oat);
    out_proj_kernel<<<1024, 256, 0, stream>>>(Oat, WoT, bo, out);
}
